// PlanarCNF_9062380995418
// MI455X (gfx1250) — matrix operand order repaired and checked by reading, not compiled, not run on hardware
//
#include <hip/hip_runtime.h>
#include <hip/hip_fp16.h>
#include <math.h>

typedef __attribute__((ext_vector_type(16))) _Float16 v16h;
typedef __attribute__((ext_vector_type(8)))  float    v8f;

#define HID 64
#define P3  640   // 3*192 + 64

struct F3 { float x, y, z; };   // 12-byte packed row of z -> global_load_b96

__device__ __forceinline__ float fast_tanhf(float x) {
#if __has_builtin(__builtin_amdgcn_tanhf)
  return __builtin_amdgcn_tanhf(x);      // v_tanh_f32 (gfx1250 transcendental)
#else
  return tanhf(x);
#endif
}

__device__ __forceinline__ float sigmoidf_dev(float x) {
  return 1.0f / (1.0f + __expf(-x));
}

// ---------------------------------------------------------------------------
// Kernel 1: hypernet  t -> (W[64x3], B[64], U[64x3] gated, wu[64]) into d_ws.
// ws layout (floats): [0,192)=W, [192,256)=B, [256,448)=U, [448,512)=wu
// ---------------------------------------------------------------------------
__global__ __launch_bounds__(256) void hypernet_kernel(
    const float* __restrict__ t,
    const float* __restrict__ fc1_w, const float* __restrict__ fc1_b,
    const float* __restrict__ fc2_w, const float* __restrict__ fc2_b,
    const float* __restrict__ fc3_w, const float* __restrict__ fc3_b,
    float* __restrict__ ws) {
  __shared__ float p1[HID];
  __shared__ float p2[HID];
  __shared__ float p3[P3];
  const int tid = threadIdx.x;

  if (tid < HID) p1[tid] = fast_tanhf(fmaf(t[0], fc1_w[tid], fc1_b[tid]));
  __syncthreads();

  if (tid < HID) {
    float acc = fc2_b[tid];
    #pragma unroll 8
    for (int k = 0; k < HID; ++k) acc = fmaf(fc2_w[tid * HID + k], p1[k], acc);
    p2[tid] = fast_tanhf(acc);
  }
  __syncthreads();

  for (int o = tid; o < P3; o += 256) {
    float acc = fc3_b[o];
    #pragma unroll 8
    for (int k = 0; k < HID; ++k) acc = fmaf(fc3_w[o * HID + k], p2[k], acc);
    p3[o] = acc;
  }
  __syncthreads();

  if (tid < 192) {
    ws[tid]       = p3[tid];                                        // W
    ws[256 + tid] = p3[192 + tid] * sigmoidf_dev(p3[384 + tid]);    // U*sig(G)
  }
  if (tid < 64) {
    ws[192 + tid] = p3[576 + tid];                                  // B
    float wu = 0.f;
    #pragma unroll
    for (int d = 0; d < 3; ++d) {
      const float u = p3[192 + tid * 3 + d] * sigmoidf_dev(p3[384 + tid * 3 + d]);
      wu = fmaf(p3[tid * 3 + d], u, wu);
    }
    ws[448 + tid] = wu;                                             // sum(W*U,axis=1)
  }
}

// ---------------------------------------------------------------------------
// Kernel 2: main batch kernel. One wave32 per 16-row tile.
//   h = tanh(z @ W^T + B)                (VALU: 3 FMA + v_tanh_f32 per unit)
//   [dz_dt | . ]      = h      @ [U|wu]  (v_wmma_f32_16x16x32_f16, K=64)
//   [ .    |trace*64] = (1-h^2)@ [U|wu]  (v_wmma_f32_16x16x32_f16, K=64)
// ---------------------------------------------------------------------------
__global__ __launch_bounds__(256) void planar_cnf_kernel(
    const float* __restrict__ z, const float* __restrict__ ws,
    float* __restrict__ out_dz, float* __restrict__ out_lp, int batch) {
  // LDS: per-k constants packed for one ds_load_b128: {W0, W1, W2, B}
  __shared__ float4 sWB[HID];
  const int tid = threadIdx.x;
  for (int k = tid; k < HID; k += 256)
    sWB[k] = make_float4(ws[k * 3 + 0], ws[k * 3 + 1], ws[k * 3 + 2], ws[192 + k]);
  __syncthreads();

  const int lane = tid & 31;
  const int half = lane >> 4;   // 0: lanes 0-15, 1: lanes 16-31
  const int n    = lane & 15;   // matrix row (A) / column (B,D) index

  // B fragments (32x16 f16), constant across tiles: col 0..2 = U, col 3 = wu.
  // Lane layout: N = n, VGPR v holds K = 32c + 16*half + {2v, 2v+1}.
  const float* Ug  = ws + 256;
  const float* wug = ws + 448;
  v16h bf[2];
  #pragma unroll
  for (int c = 0; c < 2; ++c) {
    #pragma unroll
    for (int e = 0; e < 16; ++e) {
      const int k = 32 * c + 16 * half + e;
      float v = 0.0f;
      if (n < 3)       v = Ug[k * 3 + n];
      else if (n == 3) v = wug[k];
      bf[c][e] = (_Float16)v;
    }
  }

  v16h one16;
  #pragma unroll
  for (int e = 0; e < 16; ++e) one16[e] = (_Float16)1.0f;

  const F3* __restrict__ z3 = (const F3*)z;
  const int ntiles = (batch + 15) >> 4;
  const int wave   = (int)blockIdx.x * (int)(blockDim.x >> 5) + (tid >> 5);
  const int nwaves = (int)gridDim.x * (int)(blockDim.x >> 5);
  const float inv  = 1.0f / 64.0f;

  for (int tile = wave; tile < ntiles; tile += nwaves) {
    const int rowbase = tile << 4;
    int zr = rowbase + n;                 // both half-waves load the same rows
    if (zr >= batch) zr = batch - 1;      // clamp: keep EXEC all-ones
    const F3 zv = z3[zr];                 // one global_load_b96

    v8f acc_h = {};   // h       @ [U|wu]
    v8f acc_g = {};   // (1-h^2) @ [U|wu]
    #pragma unroll
    for (int c = 0; c < 2; ++c) {
      // A layout (16-bit, 16x32): M = n, element e -> K = 32c + 16*half + e
      v16h Ah;
      #pragma unroll
      for (int e = 0; e < 16; ++e) {
        const int k = 32 * c + 16 * half + e;
        const float4 wb = sWB[k];
        const float a = fmaf(zv.x, wb.x, fmaf(zv.y, wb.y, fmaf(zv.z, wb.z, wb.w)));
        Ah[e] = (_Float16)fast_tanhf(a);
      }
      const v16h Ag = one16 - Ah * Ah;    // packed f16: v_pk_fma_f16(-h, h, 1)

      acc_h = __builtin_amdgcn_wmma_f32_16x16x32_f16(false, Ah, false, bf[c],
                                                     (short)0, acc_h, false, false);
      acc_g = __builtin_amdgcn_wmma_f32_16x16x32_f16(false, Ag, false, bf[c],
                                                     (short)0, acc_g, false, false);
    }

    // D layout: VGPR r -> row = rowbase + r + 8*half, column = n.
    const int row0 = rowbase + 8 * half;
    if (rowbase + 16 <= batch) {          // wave-uniform fast path
      if (n < 3) {
        // 8 stores with constant byte offsets 0,12,...,84 off one base pointer
        float* p = out_dz + (size_t)row0 * 3 + n;
        #pragma unroll
        for (int r = 0; r < 8; ++r) p[r * 3] = acc_h[r] * inv;
      } else if (n == 3) {
        // 8 consecutive rows of out_lp -> two b128 stores (row0 % 8 == 0)
        float4* p = (float4*)(out_lp + row0);
        p[0] = make_float4(-inv * acc_g[0], -inv * acc_g[1],
                           -inv * acc_g[2], -inv * acc_g[3]);
        p[1] = make_float4(-inv * acc_g[4], -inv * acc_g[5],
                           -inv * acc_g[6], -inv * acc_g[7]);
      }
    } else {                              // cold tail: per-row guarded
      if (n < 3) {
        float* p = out_dz + (size_t)row0 * 3 + n;
        #pragma unroll
        for (int r = 0; r < 8; ++r)
          if (row0 + r < batch) p[r * 3] = acc_h[r] * inv;
      } else if (n == 3) {
        #pragma unroll
        for (int r = 0; r < 8; ++r)
          if (row0 + r < batch) out_lp[row0 + r] = -inv * acc_g[r];
      }
    }
  }
}

// ---------------------------------------------------------------------------
extern "C" void kernel_launch(void* const* d_in, const int* in_sizes, int n_in,
                              void* d_out, int out_size, void* d_ws, size_t ws_size,
                              hipStream_t stream) {
  const float* t     = (const float*)d_in[0];
  const float* zp    = (const float*)d_in[1];
  // d_in[2] = logp_z (all zeros in reference; dlogp does not depend on it)
  const float* fc1_w = (const float*)d_in[3];
  const float* fc1_b = (const float*)d_in[4];
  const float* fc2_w = (const float*)d_in[5];
  const float* fc2_b = (const float*)d_in[6];
  const float* fc3_w = (const float*)d_in[7];
  const float* fc3_b = (const float*)d_in[8];

  const int batch = in_sizes[1] / 3;
  float* out    = (float*)d_out;
  float* out_dz = out;                       // (batch, 3) flat
  float* out_lp = out + (size_t)batch * 3;   // (batch, 1) flat
  float* ws     = (float*)d_ws;              // 512 floats used

  hypernet_kernel<<<1, 256, 0, stream>>>(t, fc1_w, fc1_b, fc2_w, fc2_b,
                                         fc3_w, fc3_b, ws);

  const int ntiles = (batch + 15) / 16;
  const int wavesPerBlock = 8;               // 256 threads = 8 wave32
  int blocks = (ntiles + wavesPerBlock - 1) / wavesPerBlock;
  if (blocks > 2048) blocks = 2048;          // grid-stride over tiles
  planar_cnf_kernel<<<blocks, 256, 0, stream>>>(zp, ws, out_dz, out_lp, batch);
}
